// Transformer_62904091018057
// MI455X (gfx1250) — matrix operand order repaired and checked by reading, not compiled, not run on hardware
//
#include <hip/hip_runtime.h>
#include <hip/hip_bf16.h>
#include <float.h>

// ---------------- problem constants (match reference) ----------------
#define NND  100000      // nodes
#define NE   800000      // edges
#define NG   1000        // graphs
#define KDIM 128         // N_IN == HC == 128
#define NOUTD 768
#define NPAIRS (NND/32)     // 3125 wave-pairs of 16-row tiles (exact)
#define GTILES 63           // ceil(1000/16)

typedef __attribute__((ext_vector_type(16))) _Float16 v16h;
typedef __attribute__((ext_vector_type(8)))  float    v8f;
typedef unsigned int u32x4 __attribute__((ext_vector_type(4)));
typedef int          i32x8 __attribute__((ext_vector_type(8)));
typedef int          i32x4 __attribute__((ext_vector_type(4)));

union V16HU { float4 f4[2]; unsigned int u32[8]; v16h v; };

// ---------------- small helpers ----------------
__device__ __forceinline__ void atomicMaxF(float* a, float val) {
  if (val >= 0.0f) atomicMax((int*)a, __float_as_int(val));
  else             atomicMin((unsigned int*)a, __float_as_uint(val));
}
__device__ __forceinline__ float leaky(float x) { return x >= 0.0f ? x : 0.01f * x; }

// Issue a TDM copy of `nbytes/2` f16 elements (contiguous) from global -> LDS.
// Descriptor per CDNA5 ISA ch.8: group0 {count=1,type=2,lds,gaddr}, group1
// {data_size=2B, 1-row tile of n elements, tensor_dim0=stride=n}.
__device__ __forceinline__ void tdm_copy_f16(unsigned int lds_addr,
                                             unsigned long long gaddr,
                                             unsigned int nelem) {
  u32x4 g0;
  g0[0] = 1u;                                    // count=1, user descriptor
  g0[1] = lds_addr;                              // LDS byte address
  g0[2] = (unsigned int)(gaddr & 0xFFFFFFFFull); // global addr [31:0]
  g0[3] = (unsigned int)((gaddr >> 32) & 0x01FFFFFFu) | (2u << 30); // [56:32] | type=2
  i32x8 g1;
  g1[0] = 0x00010000;                 // workgroup_mask=0, data_size=1 (2 bytes)
  g1[1] = (int)(nelem << 16);         // tensor_dim0 low16 -> bits[63:48]
  g1[2] = 0x00010000 | (int)(nelem >> 16); // tensor_dim0 hi16, tensor_dim1=1
  g1[3] = (int)(nelem << 16);         // tile_dim0 = nelem (bits[127:112])
  g1[4] = 1;                          // tile_dim1=1, tile_dim2=0
  g1[5] = (int)nelem;                 // tensor_dim0_stride low32
  g1[6] = 0;                          // stride hi16, dim1_stride low16
  g1[7] = 0;
  i32x4 z4 = {0, 0, 0, 0};
#if defined(__clang_major__) && (__clang_major__ >= 23)
  i32x8 z8 = {0, 0, 0, 0, 0, 0, 0, 0};
  __builtin_amdgcn_tensor_load_to_lds(g0, g1, z4, z4, z8, 0);
#else
  __builtin_amdgcn_tensor_load_to_lds(g0, g1, z4, z4, 0);
#endif
}

// ---------------- fills / conversions ----------------
__global__ void fill_f32(float* p, int n, float v) {
  int i = blockIdx.x * 256 + threadIdx.x;
  if (i < n) p[i] = v;
}

__global__ void cvt_f16(const float* __restrict__ x, _Float16* __restrict__ xh, int n) {
  int i = blockIdx.x * 256 + threadIdx.x;
  if (i < n) xh[i] = (_Float16)x[i];
}

// Swizzle the 4 [128x128] weight mats into per-lane WMMA B-operand order:
// operand op = ((mat*8+nt)*4+kc); storage: wB[op*512 + lane*16 + t]
// B element (lane,t): n = nt*16 + (lane&15); k = kc*32 + (lane>>4)*8 + (t < 8 ? t : t + 8)
__global__ void swz_w4(const float* __restrict__ Wq, const float* __restrict__ Wk,
                       const float* __restrict__ Wv, const float* __restrict__ Ws,
                       _Float16* __restrict__ wB) {
  int tid = blockIdx.x * 256 + threadIdx.x;           // 4*8*4*512 = 65536
  if (tid >= 4 * 8 * 4 * 512) return;
  int op = tid >> 9, within = tid & 511;
  int lane = within >> 4, t = within & 15;
  int mat = op >> 5, rem = op & 31, nt = rem >> 2, kc = rem & 3;
  int n  = nt * 16 + (lane & 15);
  int kk = kc * 32 + (lane >> 4) * 8 + (t < 8 ? t : t + 8);
  const float* W = (mat == 0) ? Wq : (mat == 1) ? Wk : (mat == 2) ? Wv : Ws;
  wB[tid] = (_Float16)W[kk * KDIM + n];
}

// Same swizzle for W2 [128 x 768]: 48 n-tiles x 4 k-chunks
__global__ void swz_w2(const float* __restrict__ W2, _Float16* __restrict__ w2B) {
  int tid = blockIdx.x * 256 + threadIdx.x;           // 48*4*512 = 98304
  if (tid >= 48 * 4 * 512) return;
  int op = tid >> 9, within = tid & 511;
  int lane = within >> 4, t = within & 15;
  int nt = op >> 2, kc = op & 3;
  int n  = nt * 16 + (lane & 15);
  int kk = kc * 32 + (lane >> 4) * 8 + (t < 8 ? t : t + 8);
  w2B[tid] = (_Float16)W2[kk * NOUTD + n];
}

// ------ fused QKVS GEMM: [100000x128] x 4*[128x128], 32 rows per wave --------
__global__ __launch_bounds__(128)
void gemm_qkvs(const _Float16* __restrict__ xh, const _Float16* __restrict__ wB,
               const float* __restrict__ bq, const float* __restrict__ bk,
               const float* __restrict__ bv, const float* __restrict__ bs,
               float* __restrict__ q, float* __restrict__ k,
               float* __restrict__ v, float* __restrict__ sk) {
  __shared__ __align__(16) _Float16 As[4][4096];      // 4 waves x (32 rows x 128 K)
  const int w = threadIdx.x >> 5, lane = threadIdx.x & 31;
  const int pair = blockIdx.x * 4 + w;
  const bool valid = pair < NPAIRS;
  if (valid && lane == 0) {   // TDM: async DMA of the wave's 8KB f16 tile to LDS
    unsigned int lds_base = (unsigned int)(size_t)(void*)&As[0][0] + (unsigned int)w * 8192u;
    tdm_copy_f16(lds_base, (unsigned long long)(size_t)(xh + (size_t)pair * 4096), 4096u);
  }
  __builtin_amdgcn_s_wait_tensorcnt(0);
  __syncthreads();
  if (!valid) return;

  const int mrow = lane & 15, hi = lane >> 4;
  const unsigned int* As32 = (const unsigned int*)As[w];  // packed K-pairs
  // Build 2 tiles x 4 K-chunk A operands once; reused across all 32 output tiles.
  v16h a0[4], a1[4];
  #pragma unroll
  for (int kc = 0; kc < 4; ++kc) {
    V16HU u0, u1;
    #pragma unroll
    for (int p = 0; p < 8; ++p) {
      int t = 2 * p;
      int kk = kc * 32 + 8 * hi + (t < 8 ? t : t + 8);  // ISA A 16x32 f16 layout
      u0.u32[p] = As32[mrow * 64 + (kk >> 1)];          // rows 0..15
      u1.u32[p] = As32[1024 + mrow * 64 + (kk >> 1)];   // rows 16..31
    }
    a0[kc] = u0.v; a1[kc] = u1.v;
  }

  float* dsts[4]       = { q, k, v, sk };
  const float* bias[4] = { bq, bk, bv, bs };
  const int row0 = pair * 32;
  #pragma unroll
  for (int mat = 0; mat < 4; ++mat) {
    float* D = dsts[mat];
    const float* B = bias[mat];
    #pragma unroll
    for (int nt = 0; nt < 8; ++nt) {
      v8f c0 = {0.f,0.f,0.f,0.f,0.f,0.f,0.f,0.f};
      v8f c1 = {0.f,0.f,0.f,0.f,0.f,0.f,0.f,0.f};
      #pragma unroll
      for (int kc = 0; kc < 4; ++kc) {          // each B operand feeds 2 WMMAs
        int op = ((mat * 8 + nt) * 4 + kc);
        V16HU b;
        const float4* bp = (const float4*)(wB + (size_t)op * 512 + lane * 16);
        b.f4[0] = bp[0]; b.f4[1] = bp[1];
        c0 = __builtin_amdgcn_wmma_f32_16x16x32_f16(false, a0[kc], false, b.v,
                                                    (short)0, c0, false, false);
        c1 = __builtin_amdgcn_wmma_f32_16x16x32_f16(false, a1[kc], false, b.v,
                                                    (short)0, c1, false, false);
      }
      int col = nt * 16 + mrow;
      float bia = B[col];
      #pragma unroll
      for (int r = 0; r < 8; ++r) {
        D[(size_t)(row0      + r + 8 * hi) * KDIM + col] = c0[r] + bia;
        D[(size_t)(row0 + 16 + r + 8 * hi) * KDIM + col] = c1[r] + bia;
      }
    }
  }
}

// ---------------- edge pass 1: attention logits + segment max ----------------
__global__ __launch_bounds__(256)
void edge_alpha(const int* __restrict__ ei, const float* __restrict__ q,
                const float* __restrict__ k, float* __restrict__ alpha,
                float* __restrict__ mbuf) {
  int e = blockIdx.x * 8 + (threadIdx.x >> 5);
  int lane = threadIdx.x & 31;
  if (e >= NE) return;
  int s = ei[e], d = ei[NE + e];
  float4 qv = ((const float4*)(q + (size_t)d * KDIM))[lane];   // coalesced 512B row
  float4 kv = ((const float4*)(k + (size_t)s * KDIM))[lane];
  float part = qv.x*kv.x + qv.y*kv.y + qv.z*kv.z + qv.w*kv.w;
  part += __shfl_xor(part, 1);
  part += __shfl_xor(part, 2);
  part += __shfl_xor(part, 4);          // reduce 8-lane head groups
  if ((lane & 7) == 0) {
    int h = lane >> 3;
    float al = part * 0.17677669529663687f;   // 1/sqrt(32)
    alpha[(size_t)e * 4 + h] = al;
    atomicMaxF(&mbuf[(size_t)d * 4 + h], al);
  }
}

// ---------------- edge pass 2: exp + segment sum ----------------
__global__ void edge_exp(const int* __restrict__ ei, float* __restrict__ alpha,
                         const float* __restrict__ mbuf, float* __restrict__ denom) {
  int tid = blockIdx.x * 256 + threadIdx.x;     // over E*4
  if (tid >= NE * 4) return;
  int e = tid >> 2, h = tid & 3;
  int d = ei[NE + e];
  float ex = __expf(alpha[tid] - mbuf[(size_t)d * 4 + h]);
  alpha[tid] = ex;
  atomicAdd(&denom[(size_t)d * 4 + h], ex);
}

// ---------------- edge pass 3: weighted message scatter ----------------
__global__ __launch_bounds__(256)
void edge_msg(const int* __restrict__ ei, const float* __restrict__ v,
              const float* __restrict__ alpha, const float* __restrict__ denom,
              float* __restrict__ outb) {
  int e = blockIdx.x * 8 + (threadIdx.x >> 5);
  int lane = threadIdx.x & 31;
  if (e >= NE) return;
  int s = ei[e], d = ei[NE + e];
  int h = lane >> 3;
  float attn = alpha[(size_t)e * 4 + h] / (denom[(size_t)d * 4 + h] + 1e-16f);
  float4 vv = ((const float4*)(v + (size_t)s * KDIM))[lane];
  float* o = outb + (size_t)d * KDIM + lane * 4;
  atomicAdd(o + 0, vv.x * attn);
  atomicAdd(o + 1, vv.y * attn);
  atomicAdd(o + 2, vv.z * attn);
  atomicAdd(o + 3, vv.w * attn);
}

// ---------------- node pass: beta gate + LeakyReLU + graph pool accumulate ----
__global__ __launch_bounds__(256)
void node_gate(const float* __restrict__ outb, const float* __restrict__ skip,
               const float* __restrict__ Wbeta, const int* __restrict__ batch,
               float* __restrict__ sums, float* __restrict__ cnt) {
  int nid = blockIdx.x * 8 + (threadIdx.x >> 5);
  int lane = threadIdx.x & 31;
  if (nid >= NND) return;
  float4 o = ((const float4*)(outb + (size_t)nid * KDIM))[lane];
  float4 s = ((const float4*)(skip + (size_t)nid * KDIM))[lane];
  float4 wa = ((const float4*)(Wbeta      ))[lane];
  float4 wb = ((const float4*)(Wbeta + 128))[lane];
  float4 wc = ((const float4*)(Wbeta + 256))[lane];
  float part = wa.x*o.x + wa.y*o.y + wa.z*o.z + wa.w*o.w
             + wb.x*s.x + wb.y*s.y + wb.z*s.z + wb.w*s.w
             + wc.x*(o.x-s.x) + wc.y*(o.y-s.y) + wc.z*(o.z-s.z) + wc.w*(o.w-s.w);
  #pragma unroll
  for (int off = 1; off < 32; off <<= 1) part += __shfl_xor(part, off);
  float beta = 1.0f / (1.0f + __expf(-part));
  float rx = leaky(beta*s.x + (1.0f-beta)*o.x);
  float ry = leaky(beta*s.y + (1.0f-beta)*o.y);
  float rz = leaky(beta*s.z + (1.0f-beta)*o.z);
  float rw = leaky(beta*s.w + (1.0f-beta)*o.w);
  int g = batch[nid];
  float* sg = sums + (size_t)g * KDIM + lane * 4;
  atomicAdd(sg + 0, rx);
  atomicAdd(sg + 1, ry);
  atomicAdd(sg + 2, rz);
  atomicAdd(sg + 3, rw);
  if (lane == 0) atomicAdd(&cnt[g], 1.0f);
}

// ---------------- final GEMM: pooled[1000x128] @ W2[128x768] + b2 -------------
__global__ __launch_bounds__(128)
void gemm_out(const float* __restrict__ sums, const float* __restrict__ cnt,
              const _Float16* __restrict__ w2B, const float* __restrict__ b2,
              float* __restrict__ out) {
  __shared__ __align__(16) _Float16 Ps[4][2048];
  const int w = threadIdx.x >> 5, lane = threadIdx.x & 31;
  const int tile = blockIdx.x * 4 + w;
  const bool valid = tile < GTILES;
  if (valid) {   // build pooled tile (mean pool fused) in f16
    for (int i = lane; i < 2048; i += 32) {
      int row = i >> 7, col = i & 127;
      int g = tile * 16 + row;
      float val = 0.0f;
      if (g < NG) val = sums[(size_t)g * KDIM + col] / fmaxf(cnt[g], 1.0f);
      Ps[w][i] = (_Float16)val;
    }
  }
  __syncthreads();
  if (!valid) return;

  const int mrow = lane & 15, hi = lane >> 4;
  const unsigned int* Ps32 = (const unsigned int*)Ps[w];
  v16h a[4];
  #pragma unroll
  for (int kc = 0; kc < 4; ++kc) {
    V16HU au;
    #pragma unroll
    for (int p = 0; p < 8; ++p) {
      int t = 2 * p;
      int kk = kc * 32 + 8 * hi + (t < 8 ? t : t + 8);
      au.u32[p] = Ps32[mrow * 64 + (kk >> 1)];
    }
    a[kc] = au.v;
  }
  for (int nt = 0; nt < 48; ++nt) {
    v8f c = {0.f,0.f,0.f,0.f,0.f,0.f,0.f,0.f};
    #pragma unroll
    for (int kc = 0; kc < 4; ++kc) {
      int op = nt * 4 + kc;
      V16HU b;
      const float4* bp = (const float4*)(w2B + (size_t)op * 512 + lane * 16);
      b.f4[0] = bp[0]; b.f4[1] = bp[1];
      c = __builtin_amdgcn_wmma_f32_16x16x32_f16(false, a[kc], false, b.v,
                                                 (short)0, c, false, false);
    }
    int col = nt * 16 + mrow;
    float bia = b2[col];
    #pragma unroll
    for (int r = 0; r < 8; ++r) {
      int g = tile * 16 + r + 8 * hi;
      if (g < NG) out[(size_t)g * NOUTD + col] = c[r] + bia;
    }
  }
}

// ---------------- launcher ----------------
extern "C" void kernel_launch(void* const* d_in, const int* in_sizes, int n_in,
                              void* d_out, int out_size, void* d_ws, size_t ws_size,
                              hipStream_t stream) {
  const float* x     = (const float*)d_in[0];
  const int*   ei    = (const int*)  d_in[1];   // [2,E]: src then dst
  const int*   batch = (const int*)  d_in[2];
  const float* Wq = (const float*)d_in[3];  const float* bq = (const float*)d_in[4];
  const float* Wk = (const float*)d_in[5];  const float* bk = (const float*)d_in[6];
  const float* Wv = (const float*)d_in[7];  const float* bv = (const float*)d_in[8];
  const float* Ws = (const float*)d_in[9];  const float* bs = (const float*)d_in[10];
  const float* Wbeta = (const float*)d_in[11];
  const float* W2 = (const float*)d_in[12]; const float* b2 = (const float*)d_in[13];
  float* out = (float*)d_out;

  // workspace carve-out (256B aligned regions)
  char* base = (char*)d_ws;
  size_t off = 0;
  auto carve = [&](size_t bytes) { void* p = base + off; off += (bytes + 255) & ~(size_t)255; return p; };
  _Float16* xh   = (_Float16*)carve((size_t)NND * KDIM * 2);    // x in f16
  _Float16* wB   = (_Float16*)carve(4 * 8 * 4 * 512 * 2);       // swizzled QKVS weights
  _Float16* w2B  = (_Float16*)carve(48 * 4 * 512 * 2);          // swizzled W2
  float* q    = (float*)carve((size_t)NND * KDIM * 4);
  float* k    = (float*)carve((size_t)NND * KDIM * 4);
  float* v    = (float*)carve((size_t)NND * KDIM * 4);
  float* sk   = (float*)carve((size_t)NND * KDIM * 4);
  float* outb = (float*)carve((size_t)NND * KDIM * 4);
  float* alpha= (float*)carve((size_t)NE * 4 * 4);
  float* mbuf = (float*)carve((size_t)NND * 4 * 4);
  float* denom= (float*)carve((size_t)NND * 4 * 4);
  float* sums = (float*)carve((size_t)NG * KDIM * 4);
  float* cnt  = (float*)carve((size_t)NG * 4);

  // init accumulators
  fill_f32<<<(NND*4 + 255)/256, 256, 0, stream>>>(mbuf, NND*4, -3.402823466e38f);
  fill_f32<<<(NND*4 + 255)/256, 256, 0, stream>>>(denom, NND*4, 0.0f);
  fill_f32<<<(NND*KDIM + 255)/256, 256, 0, stream>>>(outb, NND*KDIM, 0.0f);
  fill_f32<<<(NG*KDIM + 255)/256, 256, 0, stream>>>(sums, NG*KDIM, 0.0f);
  fill_f32<<<(NG + 255)/256, 256, 0, stream>>>(cnt, NG, 0.0f);

  // precision prep + weight swizzle into WMMA lane order
  cvt_f16<<<(NND*KDIM + 255)/256, 256, 0, stream>>>(x, xh, NND*KDIM);
  swz_w4<<<(4*8*4*512 + 255)/256, 256, 0, stream>>>(Wq, Wk, Wv, Ws, wB);
  swz_w2<<<(48*4*512 + 255)/256, 256, 0, stream>>>(W2, w2B);

  // fused q/k/v/skip projection (WMMA + TDM tile staging)
  gemm_qkvs<<<(NPAIRS + 3)/4, 128, 0, stream>>>(xh, wB, bq, bk, bv, bs, q, k, v, sk);

  // edge attention: logits+max, exp+sum, weighted scatter
  edge_alpha<<<(NE + 7)/8, 256, 0, stream>>>(ei, q, k, alpha, mbuf);
  edge_exp  <<<(NE*4 + 255)/256, 256, 0, stream>>>(ei, alpha, mbuf, denom);
  edge_msg  <<<(NE + 7)/8, 256, 0, stream>>>(ei, v, alpha, denom, outb);

  // gate + pool accumulate
  node_gate<<<(NND + 7)/8, 256, 0, stream>>>(outb, sk, Wbeta, batch, sums, cnt);

  // final projection (WMMA)
  gemm_out<<<(GTILES + 3)/4, 128, 0, stream>>>(sums, cnt, w2B, b2, out);
}
